// LigandContextSurrogateModel_23802708755006
// MI455X (gfx1250) — matrix operand order repaired and checked by reading, not compiled, not run on hardware
//
#include <hip/hip_runtime.h>
#include <hip/hip_bf16.h>

typedef __attribute__((ext_vector_type(16))) _Float16 v16h;
typedef __attribute__((ext_vector_type(8)))  _Float16 v8h;
typedef __attribute__((ext_vector_type(8)))  float    v8f;

#define H 128
#define NUM_RBF 32
#define EF_STRIDE 33   // 32 radial + 1 dist
#define M_IN 417
#define KP1 448        // M_IN padded to multiple of 32
#define KS_MSG1 14     // 448/32
#define KS_HH 4        // 128/32
#define KS_UPD1 8      // 256/32
#define LAYERS 4
#define ME 64          // edges per block in k_msg (4 row-tiles/wave)
#define MU 64          // nodes per block in k_upd

__device__ __forceinline__ float silu(float x) {
    return x / (1.0f + __expf(-x));
}

// Load A fragment (16x32 f16, ISA 7.12.2 A layout) from an LDS row.
// lane: row = lane%16, kb = (lane<16)?0:8; elements 0..7 -> K=kb..kb+7,
// elements 8..15 -> K=kb+16..kb+23  => two contiguous 16B chunks.
__device__ __forceinline__ v16h load_a_frag(const _Float16* row, int k0) {
    union { v16h v; v8h h[2]; } u;
    u.h[0] = *(const v8h*)(row + k0);
    u.h[1] = *(const v8h*)(row + k0 + 16);
    return u.v;
}

__device__ __forceinline__ float block_sum_128(float v, float* sbuf) {
    int tid = threadIdx.x;
    sbuf[tid] = v;
    __syncthreads();
    for (int s = 64; s > 0; s >>= 1) {
        if (tid < s) sbuf[tid] += sbuf[tid + s];
        __syncthreads();
    }
    float r = sbuf[0];
    __syncthreads();
    return r;
}

// ---------------------------------------------------------------- utilities

__global__ void k_zero(float* __restrict__ p, long long n) {
    long long i = (long long)blockIdx.x * blockDim.x + threadIdx.x;
    long long str = (long long)gridDim.x * blockDim.x;
    for (; i < n; i += str) p[i] = 0.0f;
}

__global__ void k_f2h(const float* __restrict__ s, _Float16* __restrict__ d, int n) {
    int i = blockIdx.x * 256 + threadIdx.x;
    if (i < n) d[i] = (_Float16)s[i];
}

// Pack a row-major [Kreal,128] f32 weight into per-lane WMMA B-fragment order.
// B layout (16x16x32 f16): lane -> col = nt*16 + lane%16; K = kb + e (+8 for e>=8), kb = lane<16?0:8.
// grid.x = ksteps, block = 256 (8 n-tiles x 32 lanes)
__global__ void k_pack_b(const float* __restrict__ W, _Float16* __restrict__ out, int kreal) {
    int ks = blockIdx.x;
    int nt = threadIdx.x >> 5;
    int lane = threadIdx.x & 31;
    int col = nt * 16 + (lane & 15);
    int kb = (lane < 16) ? 0 : 8;
    _Float16* o = out + (((size_t)(ks * 8 + nt)) * 32 + lane) * 16;
#pragma unroll
    for (int e = 0; e < 16; ++e) {
        int k = ks * 32 + kb + e + ((e >= 8) ? 8 : 0);
        o[e] = (k < kreal) ? (_Float16)W[(size_t)k * H + col] : (_Float16)0.0f;
    }
}

// ---------------------------------------------------------------- small MLPs

// grid = G, block = 128: tf[g] = silu(sin_emb(t) @ w1 + b1) @ w2 + b2
__global__ void k_time(const float* __restrict__ t,
                       const float* __restrict__ w1, const float* __restrict__ b1,
                       const float* __restrict__ w2, const float* __restrict__ b2,
                       float* __restrict__ tf) {
    __shared__ float emb[64];
    __shared__ float a1[128];
    int g = blockIdx.x, j = threadIdx.x;
    float tv = t[g];
    if (j < 64) {
        int hi = j & 31;
        float fr = __expf(-9.2103403719761836f / 31.0f * (float)hi);
        float arg = tv * fr;
        emb[j] = (j >= 32) ? cosf(arg) : sinf(arg);
    }
    __syncthreads();
    float s = b1[j];
    for (int k = 0; k < 64; ++k) s += emb[k] * w1[k * H + j];
    s = silu(s);
    a1[j] = s;
    __syncthreads();
    float o = b2[j];
    for (int k = 0; k < 128; ++k) o += a1[k] * w2[k * H + j];
    tf[(size_t)g * H + j] = o;
}

// grid = N, block = 128: h = LN(emb_z[z] + emb_nt[nt] + tf[batch])
__global__ void k_init(const int* __restrict__ z, const int* __restrict__ ntp,
                       const int* __restrict__ batch, const float* __restrict__ tf,
                       const float* __restrict__ emb_z, const float* __restrict__ emb_nt,
                       const float* __restrict__ g, const float* __restrict__ b,
                       float* __restrict__ h, _Float16* __restrict__ h16) {
    __shared__ float sbuf[128];
    int n = blockIdx.x, c = threadIdx.x;
    int zc = min(max(z[n], 0), 127);
    float x = emb_z[(size_t)zc * H + c] + emb_nt[(size_t)ntp[n] * H + c]
            + tf[(size_t)batch[n] * H + c];
    float mu = block_sum_128(x, sbuf) * (1.0f / 128.0f);
    float d = x - mu;
    float var = block_sum_128(d * d, sbuf) * (1.0f / 128.0f);
    float y = d * rsqrtf(var + 1e-5f) * g[c] + b[c];
    h[(size_t)n * H + c] = y;
    h16[(size_t)n * H + c] = (_Float16)y;
}

// per-edge geometry: radial[32] + dist, stored f16
__global__ void k_geom(const float* __restrict__ pos, const int* __restrict__ src,
                       const int* __restrict__ dst, _Float16* __restrict__ ef, int E) {
    int e = blockIdx.x * 256 + threadIdx.x;
    if (e >= E) return;
    int s = src[e], d = dst[e];
    float dx = pos[s * 3 + 0] - pos[d * 3 + 0];
    float dy = pos[s * 3 + 1] - pos[d * 3 + 1];
    float dz = pos[s * 3 + 2] - pos[d * 3 + 2];
    float dist = sqrtf(dx * dx + dy * dy + dz * dz);
    const float step = 6.0f / 31.0f;
    const float gamma = 1.0f / (step * step);
#pragma unroll
    for (int i = 0; i < NUM_RBF; ++i) {
        float u = dist - step * (float)i;
        ef[(size_t)e * EF_STRIDE + i] = (_Float16)__expf(-gamma * u * u);
    }
    ef[(size_t)e * EF_STRIDE + NUM_RBF] = (_Float16)dist;
}

__global__ void k_count(const int* __restrict__ dst, float* __restrict__ cnt, int E) {
    int e = blockIdx.x * 256 + threadIdx.x;
    if (e < E) atomicAdd(&cnt[dst[e]], 1.0f);
}

// ---------------------------------------------------------------- edge message GEMM

// 64 edges per block, 8 waves; each wave owns one 16-col slab and 4 row-tiles
// (B fragment loaded once per k-step, reused for 4 WMMAs).
// GEMM1: [64,448]x[448,128], silu; GEMM2: [64,128]x[128,128], silu; atomicAdd to sums[dst].
__global__ __launch_bounds__(256) void k_msg(
    const _Float16* __restrict__ h16, const _Float16* __restrict__ ef,
    const int* __restrict__ src, const int* __restrict__ dst,
    const int* __restrict__ etype, const _Float16* __restrict__ ek16,
    const _Float16* __restrict__ pw1, const float* __restrict__ b1,
    const _Float16* __restrict__ pw2, const float* __restrict__ b2,
    float* __restrict__ sums, int E) {
    __shared__ _Float16 A[ME][KP1];
    __shared__ _Float16 Mg[ME][136];
    __shared__ int src_s[ME], dst_s[ME], et_s[ME];
    int tid = threadIdx.x;
    long long e0 = (long long)blockIdx.x * ME;
    if (tid < ME) {
        long long e = e0 + tid;
        bool v = e < (long long)E;
        src_s[tid] = v ? src[e] : 0;
        dst_s[tid] = v ? dst[e] : -1;
        et_s[tid] = v ? etype[e] : 0;
    }
    __syncthreads();
    // cols [0,256): h[src] | h[dst], 16B vector chunks
    for (int idx = tid; idx < ME * 32; idx += 256) {
        int m = idx >> 5, c = idx & 31;
        v8h v = {};
        if (e0 + m < (long long)E) {
            const _Float16* p = (c < 16) ? (h16 + (size_t)src_s[m] * H + c * 8)
                                         : (h16 + (size_t)dst_s[m] * H + (c - 16) * 8);
            v = *(const v8h*)p;
        }
        *(v8h*)&A[m][c * 8] = v;
    }
    // cols [256,384): edge-type embedding
    for (int idx = tid; idx < ME * 16; idx += 256) {
        int m = idx >> 4, c = idx & 15;
        v8h v = {};
        if (e0 + m < (long long)E) v = *(const v8h*)(ek16 + (size_t)et_s[m] * H + c * 8);
        *(v8h*)&A[m][256 + c * 8] = v;
    }
    // cols [384,448): radial + dist + zero pad
    for (int idx = tid; idx < ME * 64; idx += 256) {
        int m = idx >> 6, k = 384 + (idx & 63);
        _Float16 val = (_Float16)0.0f;
        long long e = e0 + m;
        if (e < (long long)E && k < M_IN) val = ef[e * EF_STRIDE + (k - 384)];
        A[m][k] = val;
    }
    __syncthreads();

    int lane = tid & 31, w = tid >> 5;
    int r16 = lane & 15;
    int kbA = (lane < 16) ? 0 : 8;
    int mofs = (lane < 16) ? 0 : 8;
    int col = w * 16 + r16;

    v8f acc[4] = {};
#pragma unroll
    for (int ks = 0; ks < KS_MSG1; ++ks) {
        v16h bfr = *(const v16h*)(pw1 + (((size_t)(ks * 8 + w)) * 32 + lane) * 16);
#pragma unroll
        for (int ti = 0; ti < 4; ++ti) {
            v16h a = load_a_frag(&A[ti * 16 + r16][0], ks * 32 + kbA);
            acc[ti] = __builtin_amdgcn_wmma_f32_16x16x32_f16(false, a, false, bfr,
                                                             (short)0, acc[ti], false, false);
        }
    }
    float bias1 = b1[col];
#pragma unroll
    for (int ti = 0; ti < 4; ++ti)
#pragma unroll
        for (int r = 0; r < 8; ++r) {
            float x = acc[ti][r] + bias1;
            Mg[ti * 16 + r + mofs][col] = (_Float16)silu(x);
        }
    __syncthreads();

    v8f acc2[4] = {};
#pragma unroll
    for (int ks = 0; ks < KS_HH; ++ks) {
        v16h bfr = *(const v16h*)(pw2 + (((size_t)(ks * 8 + w)) * 32 + lane) * 16);
#pragma unroll
        for (int ti = 0; ti < 4; ++ti) {
            v16h a = load_a_frag(&Mg[ti * 16 + r16][0], ks * 32 + kbA);
            acc2[ti] = __builtin_amdgcn_wmma_f32_16x16x32_f16(false, a, false, bfr,
                                                              (short)0, acc2[ti], false, false);
        }
    }
    float bias2 = b2[col];
#pragma unroll
    for (int ti = 0; ti < 4; ++ti)
#pragma unroll
        for (int r = 0; r < 8; ++r) {
            int m = ti * 16 + r + mofs;
            int d = dst_s[m];
            if (d >= 0) {
                float x = silu(acc2[ti][r] + bias2);
                atomicAdd(&sums[(size_t)d * H + col], x);
            }
        }
}

// ---------------------------------------------------------------- node update GEMM

// 64 nodes per block: upd = silu([h|agg] @ w1 + b1) @ w2 + b2
__global__ __launch_bounds__(256) void k_upd(
    const _Float16* __restrict__ h16, const float* __restrict__ sums,
    const float* __restrict__ cnt,
    const _Float16* __restrict__ pw1, const float* __restrict__ b1,
    const _Float16* __restrict__ pw2, const float* __restrict__ b2,
    float* __restrict__ upd, int N) {
    __shared__ _Float16 A[MU][256];
    __shared__ _Float16 Mg[MU][136];
    __shared__ float rinv_s[MU];
    int tid = threadIdx.x;
    long long n0 = (long long)blockIdx.x * MU;
    if (tid < MU) {
        long long n = n0 + tid;
        rinv_s[tid] = (n < (long long)N) ? 1.0f / fmaxf(cnt[n], 1.0f) : 0.0f;
    }
    __syncthreads();
    // cols [0,128): h16 (16B chunks)
    for (int idx = tid; idx < MU * 16; idx += 256) {
        int m = idx >> 4, c = idx & 15;
        v8h v = {};
        long long n = n0 + m;
        if (n < (long long)N) v = *(const v8h*)(h16 + (size_t)n * H + c * 8);
        *(v8h*)&A[m][c * 8] = v;
    }
    // cols [128,256): agg = sums/cnt
    for (int idx = tid; idx < MU * H; idx += 256) {
        int m = idx >> 7, c = idx & 127;
        long long n = n0 + m;
        _Float16 val = (_Float16)0.0f;
        if (n < (long long)N) val = (_Float16)(sums[(size_t)n * H + c] * rinv_s[m]);
        A[m][128 + c] = val;
    }
    __syncthreads();

    int lane = tid & 31, w = tid >> 5;
    int r16 = lane & 15;
    int kbA = (lane < 16) ? 0 : 8;
    int mofs = (lane < 16) ? 0 : 8;
    int col = w * 16 + r16;

    v8f acc[4] = {};
#pragma unroll
    for (int ks = 0; ks < KS_UPD1; ++ks) {
        v16h bfr = *(const v16h*)(pw1 + (((size_t)(ks * 8 + w)) * 32 + lane) * 16);
#pragma unroll
        for (int ti = 0; ti < 4; ++ti) {
            v16h a = load_a_frag(&A[ti * 16 + r16][0], ks * 32 + kbA);
            acc[ti] = __builtin_amdgcn_wmma_f32_16x16x32_f16(false, a, false, bfr,
                                                             (short)0, acc[ti], false, false);
        }
    }
    float bias1 = b1[col];
#pragma unroll
    for (int ti = 0; ti < 4; ++ti)
#pragma unroll
        for (int r = 0; r < 8; ++r) {
            float x = acc[ti][r] + bias1;
            Mg[ti * 16 + r + mofs][col] = (_Float16)silu(x);
        }
    __syncthreads();

    v8f acc2[4] = {};
#pragma unroll
    for (int ks = 0; ks < KS_HH; ++ks) {
        v16h bfr = *(const v16h*)(pw2 + (((size_t)(ks * 8 + w)) * 32 + lane) * 16);
#pragma unroll
        for (int ti = 0; ti < 4; ++ti) {
            v16h a = load_a_frag(&Mg[ti * 16 + r16][0], ks * 32 + kbA);
            acc2[ti] = __builtin_amdgcn_wmma_f32_16x16x32_f16(false, a, false, bfr,
                                                              (short)0, acc2[ti], false, false);
        }
    }
    float bias2 = b2[col];
#pragma unroll
    for (int ti = 0; ti < 4; ++ti)
#pragma unroll
        for (int r = 0; r < 8; ++r) {
            long long n = n0 + ti * 16 + r + mofs;
            if (n < (long long)N) upd[(size_t)n * H + col] = acc2[ti][r] + bias2;
        }
}

// conditional LayerNorm residual: h = lig ? LN(h+upd)*g+b : h
__global__ void k_ln(float* __restrict__ h, _Float16* __restrict__ h16,
                     const float* __restrict__ upd, const int* __restrict__ ntp,
                     const float* __restrict__ g, const float* __restrict__ b) {
    __shared__ float sbuf[128];
    int n = blockIdx.x, c = threadIdx.x;
    if (ntp[n] != 1) return;  // uniform per block
    float x = h[(size_t)n * H + c] + upd[(size_t)n * H + c];
    float mu = block_sum_128(x, sbuf) * (1.0f / 128.0f);
    float d = x - mu;
    float var = block_sum_128(d * d, sbuf) * (1.0f / 128.0f);
    float y = d * rsqrtf(var + 1e-5f) * g[c] + b[c];
    h[(size_t)n * H + c] = y;
    h16[(size_t)n * H + c] = (_Float16)y;
}

// ---------------------------------------------------------------- pooling + head

__global__ void k_pool(const float* __restrict__ h, const int* __restrict__ ntp,
                       const int* __restrict__ batch,
                       float* __restrict__ gsum, float* __restrict__ gcnt) {
    int n = blockIdx.x, c = threadIdx.x;
    if (ntp[n] != 1) return;
    atomicAdd(&gsum[(size_t)batch[n] * H + c], h[(size_t)n * H + c]);
    if (c == 0) atomicAdd(&gcnt[batch[n]], 1.0f);
}

__global__ void k_head(const float* __restrict__ gsum, const float* __restrict__ gcnt,
                       const float* __restrict__ w1, const float* __restrict__ b1,
                       const float* __restrict__ w2, const float* __restrict__ b2,
                       float* __restrict__ out) {
    __shared__ float gf[128];
    __shared__ float sbuf[128];
    int g = blockIdx.x, j = threadIdx.x;
    gf[j] = gsum[(size_t)g * H + j] / fmaxf(gcnt[g], 1.0f);
    __syncthreads();
    float s = b1[j];
    for (int k = 0; k < 128; ++k) s += gf[k] * w1[k * H + j];
    s = silu(s);
    float red = block_sum_128(s * w2[j], sbuf);
    if (j == 0) out[g] = red + b2[0];
}

// ---------------------------------------------------------------- launch

extern "C" void kernel_launch(void* const* d_in, const int* in_sizes, int n_in,
                              void* d_out, int out_size, void* d_ws, size_t ws_size,
                              hipStream_t stream) {
    const int*   z        = (const int*)d_in[0];
    const int*   ntp      = (const int*)d_in[1];
    const float* pos      = (const float*)d_in[2];
    const int*   eidx     = (const int*)d_in[3];
    const int*   etype    = (const int*)d_in[4];
    const int*   batch    = (const int*)d_in[5];
    const float* t        = (const float*)d_in[6];
    const float* emb_z    = (const float*)d_in[7];
    const float* emb_nt   = (const float*)d_in[8];
    const float* in_g     = (const float*)d_in[9];
    const float* in_b     = (const float*)d_in[10];
    const float* time_w1  = (const float*)d_in[11];
    const float* time_b1  = (const float*)d_in[12];
    const float* time_w2  = (const float*)d_in[13];
    const float* time_b2  = (const float*)d_in[14];
    const float* l_ek     = (const float*)d_in[15];
    const float* l_mw1    = (const float*)d_in[16];
    const float* l_mb1    = (const float*)d_in[17];
    const float* l_mw2    = (const float*)d_in[18];
    const float* l_mb2    = (const float*)d_in[19];
    const float* l_uw1    = (const float*)d_in[20];
    const float* l_ub1    = (const float*)d_in[21];
    const float* l_uw2    = (const float*)d_in[22];
    const float* l_ub2    = (const float*)d_in[23];
    const float* l_ng     = (const float*)d_in[24];
    const float* l_nb     = (const float*)d_in[25];
    const float* head_w1  = (const float*)d_in[26];
    const float* head_b1  = (const float*)d_in[27];
    const float* head_w2  = (const float*)d_in[28];
    const float* head_b2  = (const float*)d_in[29];

    const int N = in_sizes[0];
    const int E = in_sizes[4];
    const int G = in_sizes[6];

    // bump allocator over workspace
    char* wp = (char*)d_ws;
    auto alloc = [&](size_t bytes) -> void* {
        void* p = (void*)wp;
        wp += (bytes + 255) & ~(size_t)255;
        return p;
    };
    float*     tf      = (float*)alloc((size_t)G * H * 4);
    float*     h       = (float*)alloc((size_t)N * H * 4);
    _Float16*  h16     = (_Float16*)alloc((size_t)N * H * 2);
    _Float16*  ef      = (_Float16*)alloc((size_t)E * EF_STRIDE * 2);
    float*     sums    = (float*)alloc((size_t)N * H * 4);
    float*     cnt     = (float*)alloc((size_t)N * 4);
    float*     upd     = (float*)alloc((size_t)N * H * 4);
    _Float16*  ek16    = (_Float16*)alloc((size_t)LAYERS * 2 * H * 2);
    const size_t PW_M1 = (size_t)KS_MSG1 * 8 * 32 * 16;  // halves per layer
    const size_t PW_HH = (size_t)KS_HH * 8 * 32 * 16;
    const size_t PW_U1 = (size_t)KS_UPD1 * 8 * 32 * 16;
    _Float16*  pw_m1   = (_Float16*)alloc(LAYERS * PW_M1 * 2);
    _Float16*  pw_m2   = (_Float16*)alloc(LAYERS * PW_HH * 2);
    _Float16*  pw_u1   = (_Float16*)alloc(LAYERS * PW_U1 * 2);
    _Float16*  pw_u2   = (_Float16*)alloc(LAYERS * PW_HH * 2);
    float*     gsum    = (float*)alloc((size_t)G * H * 4);
    float*     gcnt    = (float*)alloc((size_t)G * 4);

    // ---- weight packing (per layer) ----
    for (int l = 0; l < LAYERS; ++l) {
        k_pack_b<<<KS_MSG1, 256, 0, stream>>>(l_mw1 + (size_t)l * M_IN * H, pw_m1 + l * PW_M1, M_IN);
        k_pack_b<<<KS_HH,   256, 0, stream>>>(l_mw2 + (size_t)l * H * H,    pw_m2 + l * PW_HH, H);
        k_pack_b<<<KS_UPD1, 256, 0, stream>>>(l_uw1 + (size_t)l * 2 * H * H, pw_u1 + l * PW_U1, 2 * H);
        k_pack_b<<<KS_HH,   256, 0, stream>>>(l_uw2 + (size_t)l * H * H,    pw_u2 + l * PW_HH, H);
    }
    k_f2h<<<(LAYERS * 2 * H + 255) / 256, 256, 0, stream>>>(l_ek, ek16, LAYERS * 2 * H);

    // ---- input embedding ----
    k_time<<<G, 128, 0, stream>>>(t, time_w1, time_b1, time_w2, time_b2, tf);
    k_init<<<N, 128, 0, stream>>>(z, ntp, batch, tf, emb_z, emb_nt, in_g, in_b, h, h16);

    const int* src = eidx;
    const int* dst = eidx + E;
    k_geom<<<(E + 255) / 256, 256, 0, stream>>>(pos, src, dst, ef, E);
    k_zero<<<512, 256, 0, stream>>>(cnt, (long long)N);
    k_count<<<(E + 255) / 256, 256, 0, stream>>>(dst, cnt, E);

    // ---- message passing layers ----
    const int msg_blocks = (E + ME - 1) / ME;
    const int upd_blocks = (N + MU - 1) / MU;
    for (int l = 0; l < LAYERS; ++l) {
        k_zero<<<2048, 256, 0, stream>>>(sums, (long long)N * H);
        k_msg<<<msg_blocks, 256, 0, stream>>>(h16, ef, src, dst, etype,
                                              ek16 + (size_t)l * 2 * H,
                                              pw_m1 + l * PW_M1, l_mb1 + (size_t)l * H,
                                              pw_m2 + l * PW_HH, l_mb2 + (size_t)l * H,
                                              sums, E);
        k_upd<<<upd_blocks, 256, 0, stream>>>(h16, sums, cnt,
                                              pw_u1 + l * PW_U1, l_ub1 + (size_t)l * H,
                                              pw_u2 + l * PW_HH, l_ub2 + (size_t)l * H,
                                              upd, N);
        k_ln<<<N, 128, 0, stream>>>(h, h16, upd, ntp,
                                    l_ng + (size_t)l * H, l_nb + (size_t)l * H);
    }

    // ---- pooling + head ----
    k_zero<<<64, 256, 0, stream>>>(gsum, (long long)G * H);
    k_zero<<<4, 128, 0, stream>>>(gcnt, (long long)G);
    k_pool<<<N, 128, 0, stream>>>(h, ntp, batch, gsum, gcnt);
    k_head<<<G, 128, 0, stream>>>(gsum, gcnt, head_w1, head_b1, head_w2, head_b2,
                                  (float*)d_out);
}
